// BOTE_V1_ABLATION_89790586290518
// MI455X (gfx1250) — matrix operand order repaired and checked by reading, not compiled, not run on hardware
//
#include <hip/hip_runtime.h>

// ---------------------------------------------------------------------------
// Problem constants (from reference)
// ---------------------------------------------------------------------------
#define BB   128
#define SS   256
#define LL   200
#define DD   768
#define PD   50
#define FEAT 818          // D + POS_DIM
#define KP1  832          // FEAT padded to mult of 32
#define RED  400          // REDUC
#define REDP 416          // padded
#define HID  150
#define HIDP 160
#define NHEAD 640         // 4 heads * 160
#define NPOL 600          // POL * HID
#define NPOLP 608         // padded to mult of 16
#define NW   (BB*LL)      // 25600 words
#define POL  4

typedef __bf16 bf16;
typedef __attribute__((ext_vector_type(16))) __bf16 v16bf;
typedef __attribute__((ext_vector_type(8)))  __bf16 v8bf;
typedef __attribute__((ext_vector_type(8)))  float  v8f;

__device__ __forceinline__ bf16 f2bf(float f) {
    unsigned u = __builtin_bit_cast(unsigned, f);
    u += 0x7FFFu + ((u >> 16) & 1u);          // round-to-nearest-even
    unsigned short h = (unsigned short)(u >> 16);
    return __builtin_bit_cast(bf16, h);
}
__device__ __forceinline__ float bf2f(bf16 b) {
    unsigned u = ((unsigned)__builtin_bit_cast(unsigned short, b)) << 16;
    return __builtin_bit_cast(float, u);
}

// ---------------------------------------------------------------------------
// 1) Pool subwords + gather POS embedding -> H bf16 [NW][KP1]
// ---------------------------------------------------------------------------
__global__ __launch_bounds__(256)
void k_prep_h(const float* __restrict__ bert, const int* __restrict__ positions,
              const int* __restrict__ postag, const float* __restrict__ embed,
              bf16* __restrict__ H) {
    const int word = blockIdx.x;              // b*LL + l
    const int b = word / LL;
    const int p0 = positions[word * 2 + 0];
    const int p1 = positions[word * 2 + 1];
    const float inv = 1.0f / (float)(p1 - p0 + 1);
    const int tag = postag[word];
    for (int d = threadIdx.x; d < KP1; d += blockDim.x) {
        float v;
        if (d < PD) {
            v = embed[tag * PD + d];
        } else if (d < FEAT) {
            const int dd = d - PD;
            float s = 0.f;
            for (int sw = p0; sw <= p1; ++sw)
                s += bert[((size_t)b * SS + sw) * DD + dd];
            v = s * inv;
        } else {
            v = 0.f;
        }
        H[(size_t)word * KP1 + d] = f2bf(v);
    }
}

// ---------------------------------------------------------------------------
// 2) Weight packing kernels (f32 -> padded bf16 [N][Kp] row-major)
// ---------------------------------------------------------------------------
__global__ void k_pack_wr(const float* __restrict__ W, bf16* __restrict__ Wr) {
    int idx = blockIdx.x * blockDim.x + threadIdx.x;
    if (idx >= REDP * KP1) return;
    int n = idx / KP1, k = idx % KP1;
    float v = (n < RED && k < FEAT) ? W[(size_t)n * FEAT + k] : 0.f;
    Wr[idx] = f2bf(v);
}

__global__ void k_pack_wh(const float* __restrict__ W0, const float* __restrict__ W1,
                          const float* __restrict__ W2, const float* __restrict__ W3,
                          bf16* __restrict__ Wh) {
    int idx = blockIdx.x * blockDim.x + threadIdx.x;
    if (idx >= NHEAD * REDP) return;
    int n = idx / REDP, k = idx % REDP;
    int hix = n / HIDP, r = n % HIDP;
    float v = 0.f;
    if (r < HID && k < RED) {
        const float* W = (hix == 0) ? W0 : (hix == 1) ? W1 : (hix == 2) ? W2 : W3;
        v = W[(size_t)r * RED + k];
    }
    Wh[idx] = f2bf(v);
}

__global__ void k_pack_wbi(const float* __restrict__ Wbi_in, bf16* __restrict__ Wbi) {
    int idx = blockIdx.x * blockDim.x + threadIdx.x;
    if (idx >= NPOLP * HIDP) return;
    int o = idx / HIDP, h = idx % HIDP;
    float v = (o < NPOL && h < HID) ? Wbi_in[(size_t)o * (HID + 1) + h] : 0.f;
    Wbi[idx] = f2bf(v);
}

__global__ void k_pack_bias(const float* __restrict__ b0, const float* __restrict__ b1,
                            const float* __restrict__ b2, const float* __restrict__ b3,
                            const float* __restrict__ Wbi_in,
                            float* __restrict__ bias_h, float* __restrict__ bias_bi) {
    int idx = blockIdx.x * blockDim.x + threadIdx.x;
    if (idx < NHEAD) {
        int hix = idx / HIDP, r = idx % HIDP;
        float v = 0.f;
        if (r < HID) {
            const float* bp = (hix == 0) ? b0 : (hix == 1) ? b1 : (hix == 2) ? b2 : b3;
            v = bp[r];
        }
        bias_h[idx] = v;
    } else if (idx < NHEAD + NPOLP) {
        int o = idx - NHEAD;
        bias_bi[o] = (o < NPOL) ? Wbi_in[(size_t)o * (HID + 1) + HID] : 0.f;
    }
}

__global__ void k_zero_bf16(bf16* __restrict__ p, int n) {
    int idx = blockIdx.x * blockDim.x + threadIdx.x;
    if (idx < n) p[idx] = __builtin_bit_cast(bf16, (unsigned short)0);
}

// ---------------------------------------------------------------------------
// 3) Generic WMMA GEMM:  Out = epilogue( A[M,K] @ W[N,K]^T + bias )
//    Block tile 64M x 128N, 8 waves, each wave 16M x 64N (4 accumulators).
//    MODE 0: +bias(n<bias_n)        -> bf16 out[m*ldo+n]                (reduc)
//    MODE 1: +bias, relu            -> bf16 out[m*ldo+n]                (heads)
//    MODE 2: +bias, scatter reshape -> bf16 out[(m*4+p)*HIDP+h]         (affine)
//    MODE 3: batched (z), transpose -> f32 triplet[(z*LL+n)*LL+l)*4+p]  (tri)
// ---------------------------------------------------------------------------
template <int MODE>
__global__ __launch_bounds__(256)
void gemm_wmma(const bf16* __restrict__ A, size_t strideAz, int lda,
               const bf16* __restrict__ W, size_t strideWz, int ldw,
               const float* __restrict__ bias, int bias_n,
               int M, int N, int K, void* __restrict__ outv, int ldo) {
    const int lane = threadIdx.x & 31;
    const int wave = threadIdx.x >> 5;
    const int z = blockIdx.z;
    const bf16* Ab = A + strideAz * (size_t)z;
    const bf16* Wb = W + strideWz * (size_t)z;

    const int m0 = blockIdx.x * 64 + (wave & 3) * 16;
    const int nb = blockIdx.y * 128 + (wave >> 2) * 64;

    // A-fragment source row (clamped; stores are predicated later)
    int mrow = m0 + (lane & 15);
    if (mrow > M - 1) mrow = M - 1;
    const int kb8 = (lane & 16) ? 8 : 0;
    const int kb16 = (lane & 16) ? 16 : 0;
    const bf16* aptr = Ab + (size_t)mrow * lda + kb16;

    const bf16* wptr[4];
#pragma unroll
    for (int j = 0; j < 4; ++j) {
        int n = nb + 16 * j + (lane & 15);
        if (n > N - 1) n = N - 1;
        wptr[j] = Wb + (size_t)n * ldw + kb16;
    }

    v8f acc[4];
#pragma unroll
    for (int j = 0; j < 4; ++j)
#pragma unroll
        for (int r = 0; r < 8; ++r) acc[j][r] = 0.f;

    for (int k0 = 0; k0 < K; k0 += 32) {
        v16bf a;
        {
            a = *(const v16bf*)(aptr + k0);
        }
#pragma unroll
        for (int j = 0; j < 4; ++j) {
            v16bf bm = *(const v16bf*)(wptr[j] + k0);
            acc[j] = __builtin_amdgcn_wmma_f32_16x16x32_bf16(
                false, a, false, bm, (short)0, acc[j], false, false);
        }
    }

    // Epilogue (C/D layout: lane 0-15 col n, VGPR r -> row r; lane 16-31 -> row 8+r)
    const int mbase = m0 + ((lane & 16) ? 8 : 0);
#pragma unroll
    for (int j = 0; j < 4; ++j) {
        const int n = nb + 16 * j + (lane & 15);
#pragma unroll
        for (int r = 0; r < 8; ++r) {
            const int m = mbase + r;
            if (m >= M || n >= N) continue;
            float v = acc[j][r];
            if (MODE == 0) {
                if (n < bias_n) v += bias[n];
                ((bf16*)outv)[(size_t)m * ldo + n] = f2bf(v);
            } else if (MODE == 1) {
                if (n < bias_n) v += bias[n];
                v = v > 0.f ? v : 0.f;
                ((bf16*)outv)[(size_t)m * ldo + n] = f2bf(v);
            } else if (MODE == 2) {
                if (n < bias_n) v += bias[n];
                if (n < NPOL) {
                    int p = n / HID, h = n - p * HID;
                    ((bf16*)outv)[((size_t)m * POL + p) * HIDP + h] = f2bf(v);
                }
            } else { // MODE 3: m = l*4+p (k-index), n = target word
                int l = m >> 2, p = m & 3;
                ((float*)outv)[(((size_t)z * LL + n) * LL + l) * POL + p] = v;
            }
        }
    }
}

// ---------------------------------------------------------------------------
// 4) Tag heads: ap_out/op_out [B,L,5] = rep @ W_tag^T + b_tag
// ---------------------------------------------------------------------------
__global__ void k_tags(const bf16* __restrict__ heads,
                       const float* __restrict__ Wap, const float* __restrict__ bap,
                       const float* __restrict__ Wop, const float* __restrict__ bop,
                       float* __restrict__ out) {
    int idx = blockIdx.x * blockDim.x + threadIdx.x; // word*10 + sel*5 + t
    if (idx >= NW * 10) return;
    int word = idx / 10, rest = idx % 10;
    int sel = rest / 5, t = rest % 5;
    const bf16* rep = heads + (size_t)word * NHEAD + sel * HIDP;
    const float* Wt = (sel ? Wop : Wap) + t * HID;
    float s = sel ? bop[t] : bap[t];
#pragma unroll 2
    for (int h = 0; h < HID; ++h) s += bf2f(rep[h]) * Wt[h];
    out[(size_t)sel * (NW * 5) + (size_t)word * 5 + t] = s;
}

// ---------------------------------------------------------------------------
// Launch
// ---------------------------------------------------------------------------
static inline int cdiv(int a, int b) { return (a + b - 1) / b; }

extern "C" void kernel_launch(void* const* d_in, const int* in_sizes, int n_in,
                              void* d_out, int out_size, void* d_ws, size_t ws_size,
                              hipStream_t stream) {
    const float* bert      = (const float*)d_in[0];
    const int*   positions = (const int*)d_in[1];
    const int*   postag    = (const int*)d_in[2];
    const float* embed     = (const float*)d_in[3];
    const float* W_reduc   = (const float*)d_in[4];
    const float* b_reduc   = (const float*)d_in[5];
    const float* W_ap      = (const float*)d_in[6];
    const float* b_ap      = (const float*)d_in[7];
    const float* W_op      = (const float*)d_in[8];
    const float* b_op      = (const float*)d_in[9];
    const float* W_ap2     = (const float*)d_in[10];
    const float* b_ap2     = (const float*)d_in[11];
    const float* W_op2     = (const float*)d_in[12];
    const float* b_op2     = (const float*)d_in[13];
    const float* W_aptag   = (const float*)d_in[14];
    const float* b_aptag   = (const float*)d_in[15];
    const float* W_optag   = (const float*)d_in[16];
    const float* b_optag   = (const float*)d_in[17];
    const float* W_bi      = (const float*)d_in[18];

    char* ws = (char*)d_ws;
    size_t off = 0;
    auto alloc = [&](size_t bytes) {
        void* p = ws + off;
        off += (bytes + 255) & ~(size_t)255;
        return p;
    };
    bf16*  H       = (bf16*)alloc((size_t)NW * KP1 * 2);       // 42.6 MB
    bf16*  Wr      = (bf16*)alloc((size_t)REDP * KP1 * 2);     // 0.69 MB
    bf16*  Wh      = (bf16*)alloc((size_t)NHEAD * REDP * 2);   // 0.53 MB
    float* bias_h  = (float*)alloc(NHEAD * 4);
    bf16*  Wbi     = (bf16*)alloc((size_t)NPOLP * HIDP * 2);   // 0.19 MB
    float* bias_bi = (float*)alloc(NPOLP * 4);
    bf16*  reduc   = (bf16*)alloc((size_t)NW * REDP * 2);      // 21.3 MB
    bf16*  heads   = (bf16*)alloc((size_t)NW * NHEAD * 2);     // 32.8 MB
    bf16*  aff     = (bf16*)alloc((size_t)BB * LL * POL * HIDP * 2); // 32.8 MB

    float* ap_op_out = (float*)d_out;                  // [2][NW][5]
    float* trip      = (float*)d_out + 2 * NW * 5;     // [B][L][L][POL]

    // Stage 0: inputs -> bf16 operands
    k_prep_h<<<NW, 256, 0, stream>>>(bert, positions, postag, embed, H);
    k_pack_wr<<<cdiv(REDP * KP1, 256), 256, 0, stream>>>(W_reduc, Wr);
    k_pack_wh<<<cdiv(NHEAD * REDP, 256), 256, 0, stream>>>(W_ap, W_op, W_ap2, W_op2, Wh);
    k_pack_wbi<<<cdiv(NPOLP * HIDP, 256), 256, 0, stream>>>(W_bi, Wbi);
    k_pack_bias<<<cdiv(NHEAD + NPOLP, 256), 256, 0, stream>>>(
        b_ap, b_op, b_ap2, b_op2, W_bi, bias_h, bias_bi);
    k_zero_bf16<<<cdiv(BB * LL * POL * HIDP, 256), 256, 0, stream>>>(
        aff, BB * LL * POL * HIDP);

    // Stage 1: reduc = H @ W_reduc^T + b   [NW, 416] bf16
    gemm_wmma<0><<<dim3(cdiv(NW, 64), cdiv(REDP, 128), 1), 256, 0, stream>>>(
        H, 0, KP1, Wr, 0, KP1, b_reduc, RED, NW, REDP, KP1, reduc, REDP);

    // Stage 2: 4 heads fused = relu(reduc @ Wh^T + b)   [NW, 640] bf16
    gemm_wmma<1><<<dim3(cdiv(NW, 64), cdiv(NHEAD, 128), 1), 256, 0, stream>>>(
        reduc, 0, REDP, Wh, 0, REDP, bias_h, NHEAD, NW, NHEAD, REDP, heads, NHEAD);

    // Stage 3: affine = [ap_node,1] @ W_bi^T, scattered to [B, L*POL, 160] bf16
    gemm_wmma<2><<<dim3(cdiv(NW, 64), cdiv(NPOLP, 128), 1), 256, 0, stream>>>(
        heads + 2 * HIDP /*ap_node cols*/, 0, NHEAD,
        Wbi, 0, HIDP, bias_bi, NPOLP, NW, NPOLP, HIDP, aff, 0);

    // Stage 4: per-batch tri = aff_b[800,160] @ op_node_b[200,160]^T -> triplet_out
    gemm_wmma<3><<<dim3(cdiv(LL * POL, 64), cdiv(LL, 128), BB), 256, 0, stream>>>(
        aff, (size_t)LL * POL * HIDP, HIDP,
        heads + 3 * HIDP /*op_node cols*/, (size_t)LL * NHEAD, NHEAD,
        nullptr, 0, LL * POL, LL, HIDP, trip, 0);

    // Stage 5: tag outputs
    k_tags<<<cdiv(NW * 10, 256), 256, 0, stream>>>(
        heads, W_aptag, b_aptag, W_optag, b_optag, ap_op_out);
}
